// GAT_13958643712844
// MI455X (gfx1250) — matrix operand order repaired and checked by reading, not compiled, not run on hardware
//
#include <hip/hip_runtime.h>
#include <hip/hip_bf16.h>

typedef __attribute__((ext_vector_type(16))) _Float16 v16h;
typedef __attribute__((ext_vector_type(8)))  _Float16 v8h;
typedef __attribute__((ext_vector_type(8)))  float    v8f;

#define NN     4096
#define NFEAT  512
#define NHID   64
#define NCLASS 16
#define NHEADS 8
#define ALPHA  0.2f

#define LDS_PITCH 40   // f16 per LDS tile row (80B: keeps 16B align, spreads banks)

__device__ __forceinline__ float lrelu(float x) { return x > 0.f ? x : ALPHA * x; }
__device__ __forceinline__ float elu(float x)   { return x > 0.f ? x : (__expf(x) - 1.f); }

// ---------------------------------------------------------------------------
// CDNA5 async global->LDS staging (ASYNCcnt path), with portable fallback.
// Probe round 2 showed the builtin's param 0 is `int4 addrspace(1)*` (global
// source, non-const) -> signature (global int4*, lds int4*, imm off, imm cpol).
// ---------------------------------------------------------------------------
#if __has_builtin(__builtin_amdgcn_global_load_async_to_lds_b128) && \
    __has_builtin(__builtin_amdgcn_s_wait_asynccnt)
#define HAVE_ASYNC_LDS 1
#else
#define HAVE_ASYNC_LDS 0
#endif

typedef __attribute__((ext_vector_type(4))) int i4_t;
typedef __attribute__((address_space(1))) i4_t* gbl_i4p;
typedef __attribute__((address_space(3))) i4_t* lds_i4p;

// copy 16 bytes (8 f16) global -> LDS
__device__ __forceinline__ void copy16_g2l(_Float16* l, const _Float16* g) {
#if HAVE_ASYNC_LDS
    __builtin_amdgcn_global_load_async_to_lds_b128(
        (gbl_i4p)(uintptr_t)g, (lds_i4p)(uintptr_t)l, 0, 0);
#else
    *(v8h*)l = *(const v8h*)g;
#endif
}

// exp-weight helper for attention A tiles
#define EW(tv) ((_Float16)__expf(lrelu(si + (tv)) - mi))

// ---------------------------------------------------------------------------
// Prep: transpose + f16-convert weights so GEMM B-operands are contiguous
// 32B loads in the WMMA dense-B layout.
//   Whk[h][f][k] = Wh[h][k][f]   (8 x 64 x 512)
//   Wot[c][k]    = Wo[k][c]      (16 x 512)
// ---------------------------------------------------------------------------
__global__ void prep_weights(const float* __restrict__ Wh, const float* __restrict__ Wo,
                             _Float16* __restrict__ Whk, _Float16* __restrict__ Wot) {
    int idx = blockIdx.x * blockDim.x + threadIdx.x;
    const int nWhk = NHEADS * NHID * NFEAT;
    if (idx < nWhk) {
        int h = idx / (NHID * NFEAT);
        int r = idx % (NHID * NFEAT);
        int f = r / NFEAT;
        int k = r % NFEAT;
        Whk[idx] = (_Float16)Wh[(h * NFEAT + k) * NHID + f];
    } else if (idx < nWhk + NCLASS * NFEAT) {
        int j = idx - nWhk;
        int c = j / NFEAT;
        int k = j % NFEAT;
        Wot[j] = (_Float16)Wo[k * NCLASS + c];
    }
}

// ---------------------------------------------------------------------------
// GEMM1: H[h] = x @ Wh[h].  One wave per (head, 16-row strip), 4 N-tiles.
// Emits H_f32 [8][4096][64] and Ht_f16 [8][64][4096] (B-operand-ready).
// ---------------------------------------------------------------------------
__global__ __launch_bounds__(256) void gemm1(const float* __restrict__ x,
                                             const _Float16* __restrict__ Whk,
                                             float* __restrict__ H,
                                             _Float16* __restrict__ Ht) {
    int wid  = (blockIdx.x * blockDim.x + threadIdx.x) >> 5;
    int lane = threadIdx.x & 31;
    int head = wid >> 8;
    int rb   = (wid & 255) << 4;
    int m    = lane & 15;
    int half = lane >> 4;
    int kb   = half * 8;

    v8f acc0 = {}, acc1 = {}, acc2 = {}, acc3 = {};
    const float* xrow = x + (size_t)(rb + m) * NFEAT;

    for (int kc = 0; kc < NFEAT; kc += 32) {
        v16h a;
#pragma unroll
        for (int v = 0; v < 8; ++v) {
            int k0 = kc + half * 16 + 2 * v;
            a[2 * v]     = (_Float16)xrow[k0];
            a[2 * v + 1] = (_Float16)xrow[k0 + 1];
        }
        const _Float16* bbase = Whk + (size_t)head * NHID * NFEAT + (size_t)(kc + half * 16);
        v16h b0 = *(const v16h*)(bbase + (size_t)(0 * 16 + m) * NFEAT);
        v16h b1 = *(const v16h*)(bbase + (size_t)(1 * 16 + m) * NFEAT);
        v16h b2 = *(const v16h*)(bbase + (size_t)(2 * 16 + m) * NFEAT);
        v16h b3 = *(const v16h*)(bbase + (size_t)(3 * 16 + m) * NFEAT);
        acc0 = __builtin_amdgcn_wmma_f32_16x16x32_f16(false, a, false, b0, (short)0, acc0, false, false);
        acc1 = __builtin_amdgcn_wmma_f32_16x16x32_f16(false, a, false, b1, (short)0, acc1, false, false);
        acc2 = __builtin_amdgcn_wmma_f32_16x16x32_f16(false, a, false, b2, (short)0, acc2, false, false);
        acc3 = __builtin_amdgcn_wmma_f32_16x16x32_f16(false, a, false, b3, (short)0, acc3, false, false);
    }

#pragma unroll
    for (int r = 0; r < 8; ++r) {
        int row = rb + half * 8 + r;
        float f0 = acc0[r], f1 = acc1[r], f2 = acc2[r], f3 = acc3[r];
        float* Hp = H + ((size_t)head * NN + row) * NHID;
        Hp[ 0 + m] = f0;  Hp[16 + m] = f1;  Hp[32 + m] = f2;  Hp[48 + m] = f3;
        _Float16* Tp = Ht + (size_t)head * NHID * NN + row;
        Tp[(size_t)( 0 + m) * NN] = (_Float16)f0;
        Tp[(size_t)(16 + m) * NN] = (_Float16)f1;
        Tp[(size_t)(32 + m) * NN] = (_Float16)f2;
        Tp[(size_t)(48 + m) * NN] = (_Float16)f3;
    }
}

// s[h][n] = H[h][n] . ah[h][:64] ;  t[h][n] = H[h][n] . ah[h][64:]
__global__ void dots1(const float* __restrict__ H, const float* __restrict__ ah,
                      float* __restrict__ s, float* __restrict__ t) {
    int idx = blockIdx.x * blockDim.x + threadIdx.x;
    if (idx >= NHEADS * NN) return;
    int head = idx >> 12;
    const float* hp = H + (size_t)idx * NHID;
    const float* a0 = ah + head * 2 * NHID;
    float ss = 0.f, tt = 0.f;
#pragma unroll 8
    for (int f = 0; f < NHID; ++f) { float hv = hp[f]; ss += hv * a0[f]; tt += hv * a0[NHID + f]; }
    s[idx] = ss; t[idx] = tt;
}

// per-segment max of t (segment length 4096); one block per segment
__global__ void colmax(const float* __restrict__ t, float* __restrict__ tmax) {
    __shared__ float red[256];
    const float* p = t + (size_t)blockIdx.x * NN;
    float mx = -3.4e38f;
    for (int j = threadIdx.x; j < NN; j += 256) mx = fmaxf(mx, p[j]);
    red[threadIdx.x] = mx; __syncthreads();
    for (int ss = 128; ss > 0; ss >>= 1) {
        if (threadIdx.x < ss) red[threadIdx.x] = fmaxf(red[threadIdx.x], red[threadIdx.x + ss]);
        __syncthreads();
    }
    if (threadIdx.x == 0) tmax[blockIdx.x] = red[0];
}

// Z[row] = sum_j exp(LR(s_i + t_j) - m_i);  one wave per row
__global__ void zsum(const float* __restrict__ s, const float* __restrict__ t,
                     const float* __restrict__ tmax, float* __restrict__ Z, int nseg) {
    int wid  = (blockIdx.x * blockDim.x + threadIdx.x) >> 5;
    int lane = threadIdx.x & 31;
    if (wid >= nseg * NN) return;
    int seg = wid >> 12;
    float si = s[wid];
    float mi = lrelu(si + tmax[seg]);
    const float* tp = t + ((size_t)seg << 12);
    float z = 0.f;
    for (int j = lane; j < NN; j += 32) z += __expf(lrelu(si + tp[j]) - mi);
#pragma unroll
    for (int off = 16; off > 0; off >>= 1) z += __shfl_xor(z, off, 32);
    if (lane == 0) Z[wid] = z;
}

// ---------------------------------------------------------------------------
// Attention layer 1: feat = softmax(e) @ H fused.  Block = 128 threads =
// 4 waves, ALL on the same head, consecutive 16-row strips.  The shared
// 64x32 f16 B tile per K-chunk is staged once per block into LDS with the
// CDNA5 async global->LDS path (double buffered, ASYNCcnt-drained), then each
// wave feeds it to 4 chained WMMAs against a register-built exp-weight A tile.
// ---------------------------------------------------------------------------
__global__ __launch_bounds__(128) void attn1(const _Float16* __restrict__ Ht,
                                             const float* __restrict__ s,
                                             const float* __restrict__ t,
                                             const float* __restrict__ tmax,
                                             const float* __restrict__ Z,
                                             _Float16* __restrict__ xc) {
    __shared__ _Float16 tile[2 * NHID * LDS_PITCH];   // 2 x 64 x 40 f16

    int tid  = threadIdx.x;            // 0..127
    int lane = tid & 31;
    int w    = tid >> 5;               // wave in block
    int head = blockIdx.x >> 6;        // 64 blocks per head
    int rb   = (((blockIdx.x & 63) << 2) + w) << 4;
    int m    = lane & 15;
    int half = lane >> 4;
    int kb   = half * 16;

    float si = s[(head << 12) + rb + m];
    float mi = lrelu(si + tmax[head]);
    const float*    tp  = t  + ((size_t)head << 12);
    const _Float16* HtH = Ht + (size_t)head * NHID * NN;

    // staging assignment: thread copies 32B of the 64x32 tile (row f, 16-f16 seg)
    int sf   = tid >> 1;
    int sseg = (tid & 1) * 16;
    const _Float16* sg = HtH + (size_t)sf * NN + sseg;
    _Float16*       sl0 = &tile[sf * LDS_PITCH + sseg];
    _Float16*       sl1 = &tile[NHID * LDS_PITCH + sf * LDS_PITCH + sseg];

    // prologue: stage chunk 0 into buffer 0
    copy16_g2l(sl0, sg);
    copy16_g2l(sl0 + 8, sg + 8);

    v8f acc0 = {}, acc1 = {}, acc2 = {}, acc3 = {};
    for (int jc = 0; jc < NN; jc += 32) {
        int cur   = (jc >> 5) & 1;
        bool more = (jc + 32) < NN;
        if (more) {                          // stage next chunk into other buffer
            const _Float16* g = sg + jc + 32;
            _Float16* l = cur ? sl0 : sl1;
            copy16_g2l(l, g);
            copy16_g2l(l + 8, g + 8);
        }
        // build exp-weight A tile (overlaps with in-flight async loads)
        float4 ta = *(const float4*)(tp + jc + kb);
        float4 tb = *(const float4*)(tp + jc + kb + 4);
        float4 tc = *(const float4*)(tp + jc + kb + 8);
        float4 td = *(const float4*)(tp + jc + kb + 12);
        v16h a;
        a[0]=EW(ta.x);  a[1]=EW(ta.y);  a[2]=EW(ta.z);  a[3]=EW(ta.w);
        a[4]=EW(tb.x);  a[5]=EW(tb.y);  a[6]=EW(tb.z);  a[7]=EW(tb.w);
        a[8]=EW(tc.x);  a[9]=EW(tc.y);  a[10]=EW(tc.z); a[11]=EW(tc.w);
        a[12]=EW(td.x); a[13]=EW(td.y); a[14]=EW(td.z); a[15]=EW(td.w);

#if HAVE_ASYNC_LDS
        if (more) __builtin_amdgcn_s_wait_asynccnt(2);   // oldest 2 (cur buffer) done
        else      __builtin_amdgcn_s_wait_asynccnt(0);
#endif
        __syncthreads();                     // cur buffer visible to all waves

        const _Float16* lb = &tile[cur * NHID * LDS_PITCH + half * 16];
        v16h b0 = *(const v16h*)(lb + ( 0 + m) * LDS_PITCH);
        v16h b1 = *(const v16h*)(lb + (16 + m) * LDS_PITCH);
        v16h b2 = *(const v16h*)(lb + (32 + m) * LDS_PITCH);
        v16h b3 = *(const v16h*)(lb + (48 + m) * LDS_PITCH);
        acc0 = __builtin_amdgcn_wmma_f32_16x16x32_f16(false, a, false, b0, (short)0, acc0, false, false);
        acc1 = __builtin_amdgcn_wmma_f32_16x16x32_f16(false, a, false, b1, (short)0, acc1, false, false);
        acc2 = __builtin_amdgcn_wmma_f32_16x16x32_f16(false, a, false, b2, (short)0, acc2, false, false);
        acc3 = __builtin_amdgcn_wmma_f32_16x16x32_f16(false, a, false, b3, (short)0, acc3, false, false);

        __syncthreads();                     // all reads of cur done before re-stage
    }

#pragma unroll
    for (int r = 0; r < 8; ++r) {
        int row = rb + half * 8 + r;
        float inv = 1.f / Z[(head << 12) + row];
        _Float16* xp = xc + (size_t)row * (NHEADS * NHID) + head * NHID;
        xp[ 0 + m] = (_Float16)elu(acc0[r] * inv);
        xp[16 + m] = (_Float16)elu(acc1[r] * inv);
        xp[32 + m] = (_Float16)elu(acc2[r] * inv);
        xp[48 + m] = (_Float16)elu(acc3[r] * inv);
    }
}

// ---------------------------------------------------------------------------
// GEMM2: G = xc @ Wo (4096x512 @ 512x16).  One wave per 16-row strip.
// ---------------------------------------------------------------------------
__global__ __launch_bounds__(256) void gemm2(const _Float16* __restrict__ xc,
                                             const _Float16* __restrict__ Wot,
                                             float* __restrict__ G,
                                             _Float16* __restrict__ Gt) {
    int wid  = (blockIdx.x * blockDim.x + threadIdx.x) >> 5;
    int lane = threadIdx.x & 31;
    int rb   = wid << 4;
    int m    = lane & 15;
    int half = lane >> 4;
    int kb   = half * 8;

    const _Float16* xrow = xc + (size_t)(rb + m) * NFEAT;
    v8f acc = {};
    for (int kc = 0; kc < NFEAT; kc += 32) {
        v16h a;
#pragma unroll
        for (int v = 0; v < 8; ++v) {
            int k0 = kc + half * 16 + 2 * v;
            a[2 * v]     = xrow[k0];
            a[2 * v + 1] = xrow[k0 + 1];
        }
        v16h b = *(const v16h*)(Wot + (size_t)m * NFEAT + kc + half * 16);
        acc = __builtin_amdgcn_wmma_f32_16x16x32_f16(false, a, false, b, (short)0, acc, false, false);
    }
#pragma unroll
    for (int r = 0; r < 8; ++r) {
        int row = rb + half * 8 + r;
        float f = acc[r];
        G[(size_t)row * NCLASS + m] = f;
        Gt[(size_t)m * NN + row]    = (_Float16)f;
    }
}

__global__ void dots2(const float* __restrict__ G, const float* __restrict__ ao,
                      float* __restrict__ s2, float* __restrict__ t2) {
    int n = blockIdx.x * blockDim.x + threadIdx.x;
    if (n >= NN) return;
    const float* gp = G + (size_t)n * NCLASS;
    float ss = 0.f, tt = 0.f;
#pragma unroll
    for (int c = 0; c < NCLASS; ++c) { float g = gp[c]; ss += g * ao[c]; tt += g * ao[NCLASS + c]; }
    s2[n] = ss; t2[n] = tt;
}

// ---------------------------------------------------------------------------
// Attention layer 2 (single "head", N=16) + outer ELU.  Same async-LDS
// double-buffered staging; the 16x32 B tile (1KB) is staged by 64 threads.
// ---------------------------------------------------------------------------
__global__ __launch_bounds__(128) void attn2(const _Float16* __restrict__ Gt,
                                             const float* __restrict__ s2,
                                             const float* __restrict__ t2,
                                             const float* __restrict__ tmax2,
                                             const float* __restrict__ Z2,
                                             float* __restrict__ feat2) {
    __shared__ _Float16 tile[2 * NCLASS * LDS_PITCH];   // 2 x 16 x 40 f16

    int tid  = threadIdx.x;
    int lane = tid & 31;
    int w    = tid >> 5;
    int rb   = ((blockIdx.x << 2) + w) << 4;
    int m    = lane & 15;
    int half = lane >> 4;
    int kb   = half * 16;

    float si = s2[rb + m];
    float mi = lrelu(si + tmax2[0]);

    bool stager = tid < 64;                 // 64 threads x 16B = 1KB tile
    int  sf     = tid >> 2;                 // row 0..15
    int  sseg   = (tid & 3) * 8;            // 8-f16 segment
    const _Float16* sg  = Gt + (size_t)sf * NN + sseg;
    _Float16*       sl0 = &tile[sf * LDS_PITCH + sseg];
    _Float16*       sl1 = &tile[NCLASS * LDS_PITCH + sf * LDS_PITCH + sseg];

    if (stager) copy16_g2l(sl0, sg);

    v8f acc = {};
    for (int jc = 0; jc < NN; jc += 32) {
        int cur   = (jc >> 5) & 1;
        bool more = (jc + 32) < NN;
        if (more && stager) copy16_g2l(cur ? sl0 : sl1, sg + jc + 32);

        float4 ta = *(const float4*)(t2 + jc + kb);
        float4 tb = *(const float4*)(t2 + jc + kb + 4);
        float4 tc = *(const float4*)(t2 + jc + kb + 8);
        float4 td = *(const float4*)(t2 + jc + kb + 12);
        v16h a;
        a[0]=EW(ta.x);  a[1]=EW(ta.y);  a[2]=EW(ta.z);  a[3]=EW(ta.w);
        a[4]=EW(tb.x);  a[5]=EW(tb.y);  a[6]=EW(tb.z);  a[7]=EW(tb.w);
        a[8]=EW(tc.x);  a[9]=EW(tc.y);  a[10]=EW(tc.z); a[11]=EW(tc.w);
        a[12]=EW(td.x); a[13]=EW(td.y); a[14]=EW(td.z); a[15]=EW(td.w);

#if HAVE_ASYNC_LDS
        if (more) __builtin_amdgcn_s_wait_asynccnt(1);
        else      __builtin_amdgcn_s_wait_asynccnt(0);
#endif
        __syncthreads();

        v16h b = *(const v16h*)(&tile[cur * NCLASS * LDS_PITCH + m * LDS_PITCH + half * 16]);
        acc = __builtin_amdgcn_wmma_f32_16x16x32_f16(false, a, false, b, (short)0, acc, false, false);

        __syncthreads();
    }

#pragma unroll
    for (int r = 0; r < 8; ++r) {
        int row = rb + half * 8 + r;
        float f = acc[r] / Z2[row];
        feat2[(size_t)row * NCLASS + m] = elu(f);
    }
}

__global__ void logsoftmax(const float* __restrict__ feat2, float* __restrict__ out) {
    int n = blockIdx.x * blockDim.x + threadIdx.x;
    if (n >= NN) return;
    const float* p = feat2 + (size_t)n * NCLASS;
    float mx = p[0];
#pragma unroll
    for (int c = 1; c < NCLASS; ++c) mx = fmaxf(mx, p[c]);
    float sum = 0.f;
#pragma unroll
    for (int c = 0; c < NCLASS; ++c) sum += __expf(p[c] - mx);
    float lse = mx + __logf(sum);
#pragma unroll
    for (int c = 0; c < NCLASS; ++c) out[(size_t)n * NCLASS + c] = p[c] - lse;
}

// ---------------------------------------------------------------------------
extern "C" void kernel_launch(void* const* d_in, const int* in_sizes, int n_in,
                              void* d_out, int out_size, void* d_ws, size_t ws_size,
                              hipStream_t stream) {
    (void)in_sizes; (void)n_in; (void)out_size; (void)ws_size;
    const float* x  = (const float*)d_in[0];
    const float* Wh = (const float*)d_in[1];
    const float* ah = (const float*)d_in[2];
    const float* Wo = (const float*)d_in[3];
    const float* ao = (const float*)d_in[4];
    float* out = (float*)d_out;

    char* ws = (char*)d_ws;
    size_t off = 0;
    auto take = [&](size_t bytes) -> char* {
        char* p = ws + off;
        off = (off + bytes + 255) & ~(size_t)255;
        return p;
    };
    float*    H     = (float*)   take((size_t)NHEADS * NN * NHID * 4);
    _Float16* Ht    = (_Float16*)take((size_t)NHEADS * NHID * NN * 2);
    _Float16* Whk   = (_Float16*)take((size_t)NHEADS * NHID * NFEAT * 2);
    _Float16* Wot   = (_Float16*)take((size_t)NCLASS * NFEAT * 2);
    float*    s1    = (float*)   take((size_t)NHEADS * NN * 4);
    float*    t1    = (float*)   take((size_t)NHEADS * NN * 4);
    float*    tmax1 = (float*)   take((size_t)NHEADS * 4);
    float*    Z1    = (float*)   take((size_t)NHEADS * NN * 4);
    _Float16* xc    = (_Float16*)take((size_t)NN * NHEADS * NHID * 2);
    float*    G     = (float*)   take((size_t)NN * NCLASS * 4);
    _Float16* Gt    = (_Float16*)take((size_t)NCLASS * NN * 2);
    float*    s2    = (float*)   take((size_t)NN * 4);
    float*    t2    = (float*)   take((size_t)NN * 4);
    float*    tmax2 = (float*)   take((size_t)4);
    float*    Z2    = (float*)   take((size_t)NN * 4);
    float*    feat2 = (float*)   take((size_t)NN * NCLASS * 4);

    {
        int total = NHEADS * NHID * NFEAT + NCLASS * NFEAT;
        prep_weights<<<(total + 255) / 256, 256, 0, stream>>>(Wh, Wo, Whk, Wot);
    }
    gemm1<<<(NHEADS * (NN / 16) * 32) / 256, 256, 0, stream>>>(x, Whk, H, Ht);
    dots1<<<(NHEADS * NN + 255) / 256, 256, 0, stream>>>(H, ah, s1, t1);
    colmax<<<NHEADS, 256, 0, stream>>>(t1, tmax1);
    zsum<<<(NHEADS * NN * 32) / 256, 256, 0, stream>>>(s1, t1, tmax1, Z1, NHEADS);
    attn1<<<NHEADS * 64, 128, 0, stream>>>(Ht, s1, t1, tmax1, Z1, xc);

    gemm2<<<((NN / 16) * 32) / 256, 256, 0, stream>>>(xc, Wot, G, Gt);
    dots2<<<(NN + 255) / 256, 256, 0, stream>>>(G, ao, s2, t2);
    colmax<<<1, 256, 0, stream>>>(t2, tmax2);
    zsum<<<(NN * 32) / 256, 256, 0, stream>>>(s2, t2, tmax2, Z2, 1);
    attn2<<<64, 128, 0, stream>>>(Gt, s2, t2, tmax2, Z2, feat2);
    logsoftmax<<<(NN + 255) / 256, 256, 0, stream>>>(feat2, out);
}
